// StandardAttention_31233002176613
// MI455X (gfx1250) — matrix operand order repaired and checked by reading, not compiled, not run on hardware
//
#include <hip/hip_runtime.h>

// ---------------- problem dims (B=1) ----------------
static constexpr int T  = 4096;
static constexpr int D  = 1024;
static constexpr int H  = 16;
static constexpr int DH = 64;          // head dim
static constexpr int N3 = 3 * D;       // 3072 (qkv row stride)

using h16 = _Float16;
typedef __attribute__((ext_vector_type(16))) h16   v16h;
typedef __attribute__((ext_vector_type(8)))  float v8f;

// D = A(16x32 f16) * B(32x16 f16) + C(16x16 f32)
static __device__ __forceinline__ v8f wmma16(v16h a, v16h b, v8f c) {
  return __builtin_amdgcn_wmma_f32_16x16x32_f16(false, a, false, b,
                                                (short)0, c, false, false);
}

// ---- CDNA5 async copy: global -> LDS, 16B per lane, tracked by ASYNCcnt ----
static __device__ __forceinline__ void async_copy16(void* lds_dst, const void* gsrc) {
  // Generic shared pointers carry the LDS byte offset in the low 32 bits.
  const unsigned lds = (unsigned)(unsigned long long)lds_dst;
  const unsigned long long ga = (unsigned long long)gsrc;
  asm volatile("global_load_async_to_lds_b128 %0, %1, off"
               :: "v"(lds), "v"(ga) : "memory");
}
#define WAIT_ASYNC_0() asm volatile("s_wait_asynccnt 0x0" ::: "memory")
#define WAIT_ASYNC_4() asm volatile("s_wait_asynccnt 0x4" ::: "memory")
#define WAIT_ASYNC_8() asm volatile("s_wait_asynccnt 0x8" ::: "memory")

// ---- fragment loaders (contiguous k = 16*(lane>=16) + s, same for A and B, wave32) ----
// A 16x32: row m = lane&15; vgpr i holds K pair; K = 16*(lane>=16) + 2*i
static __device__ __forceinline__ v16h load_a_frag(const h16* base, int ld, int lane) {
  v16h a;
  const int m  = lane & 15;
  const int kb = (lane >> 4) * 16;
#pragma unroll
  for (int i = 0; i < 8; ++i) {
    const int k = kb + 2 * i;
    a[2 * i]     = base[m * ld + k];
    a[2 * i + 1] = base[m * ld + k + 1];
  }
  return a;
}
// B 32x16 (row-major KxN source): col n = lane&15; K group = 16*(lane>=16)
static __device__ __forceinline__ v16h load_b_frag(const h16* base, int ld, int lane) {
  v16h b;
  const int n  = lane & 15;
  const int kg = (lane >> 4) * 16;
#pragma unroll
  for (int i = 0; i < 8; ++i) {
    b[2 * i]     = base[(kg + 2 * i) * ld + n];
    b[2 * i + 1] = base[(kg + 2 * i + 1) * ld + n];
  }
  return b;
}
// B 32x16 with N-major source (src[n][k]): used for Q*K^T
static __device__ __forceinline__ v16h load_b_fragT(const h16* base, int ld, int lane) {
  v16h b;
  const int n  = lane & 15;
  const int kg = (lane >> 4) * 16;
#pragma unroll
  for (int i = 0; i < 8; ++i) {
    b[2 * i]     = base[n * ld + kg + 2 * i];
    b[2 * i + 1] = base[n * ld + kg + 2 * i + 1];
  }
  return b;
}

// =====================================================================
// f32 -> f16 conversion pass (bandwidth-trivial: ~33MB total)
// =====================================================================
__global__ __launch_bounds__(256) void f32_to_f16(const float* __restrict__ src,
                                                  h16* __restrict__ dst, int n4) {
  const int i = blockIdx.x * 256 + threadIdx.x;
  if (i < n4) {
    const float4 v = reinterpret_cast<const float4*>(src)[i];
    union { h16 h[4]; uint2 u; } t;
    t.h[0] = (h16)v.x; t.h[1] = (h16)v.y; t.h[2] = (h16)v.z; t.h[3] = (h16)v.w;
    reinterpret_cast<uint2*>(dst)[i] = t.u;
  }
}

// =====================================================================
// Tiled WMMA GEMM (f16 x f16): C[MxN] = A[MxK] * B[KxN]
// 128x128x32 tiles, 256 thr = 8 waves (2x4), wave tile 64x32 = 4x2 WMMA.
// Double-buffered LDS filled with async global->LDS copies (ASYNCcnt).
// =====================================================================
static constexpr int GLDA = 40;   // 32 + 8 pad (halves)
static constexpr int GLDB = 136;  // 128 + 8 pad

static __device__ __forceinline__ void stage_gemm_tile(
    h16 (*As)[GLDA], h16 (*Bs)[GLDB], const h16* __restrict__ Ah,
    const h16* __restrict__ Bh, int m0, int n0, int k0, int K, int N, int tid) {
  // A tile 128x32 halves: 16B chunks, 4 chunks/row, 2 passes  (2 async instr)
#pragma unroll
  for (int p = 0; p < 2; ++p) {
    const int r = p * 64 + (tid >> 2);
    const int c = (tid & 3) * 8;
    async_copy16(&As[r][c], Ah + (size_t)(m0 + r) * K + k0 + c);
  }
  // B tile 32x128 halves: 16 chunks/row, 2 passes               (2 async instr)
#pragma unroll
  for (int p = 0; p < 2; ++p) {
    const int r = p * 16 + (tid >> 4);
    const int c = (tid & 15) * 8;
    async_copy16(&Bs[r][c], Bh + (size_t)(k0 + r) * N + n0 + c);
  }
}

template <bool OUT_F16>
__global__ __launch_bounds__(256) void gemm_wmma(const h16* __restrict__ Ah,
                                                 const h16* __restrict__ Bh,
                                                 void* __restrict__ Cp,
                                                 int M, int N, int K) {
  (void)M;
  constexpr int BM = 128, BN = 128, BK = 32;
  __shared__ h16 As[2][BM][GLDA];
  __shared__ h16 Bs[2][BK][GLDB];

  const int tid  = threadIdx.x;
  const int lane = tid & 31;
  const int wave = tid >> 5;
  const int wm   = wave >> 2;  // 0..1
  const int wn   = wave & 3;   // 0..3
  const int m0   = blockIdx.y * BM;
  const int n0   = blockIdx.x * BN;

  const v8f vzero = {};
  v8f acc[4][2];
#pragma unroll
  for (int mt = 0; mt < 4; ++mt)
#pragma unroll
    for (int nt = 0; nt < 2; ++nt) acc[mt][nt] = vzero;

  stage_gemm_tile(As[0], Bs[0], Ah, Bh, m0, n0, 0, K, N, tid);

  int buf = 0;
  for (int k0 = 0; k0 < K; k0 += BK) {
    if (k0 + BK < K) {
      stage_gemm_tile(As[buf ^ 1], Bs[buf ^ 1], Ah, Bh, m0, n0, k0 + BK, K, N, tid);
      WAIT_ASYNC_4();           // current tile's 4 async ops retired
    } else {
      WAIT_ASYNC_0();
    }
    __syncthreads();            // tile visible to all waves

    v16h aF[4];
#pragma unroll
    for (int mt = 0; mt < 4; ++mt)
      aF[mt] = load_a_frag(&As[buf][wm * 64 + mt * 16][0], GLDA, lane);
    v16h bF[2];
#pragma unroll
    for (int nt = 0; nt < 2; ++nt)
      bF[nt] = load_b_frag(&Bs[buf][0][wn * 32 + nt * 16], GLDB, lane);
#pragma unroll
    for (int mt = 0; mt < 4; ++mt)
#pragma unroll
      for (int nt = 0; nt < 2; ++nt)
        acc[mt][nt] = wmma16(aF[mt], bF[nt], acc[mt][nt]);

    __syncthreads();            // all waves done reading buf before overwrite
    buf ^= 1;
  }

  // ---- epilogue: C/D layout M = r + 8*(lane>=16), N = lane&15 ----
  const int hi = (lane >> 4) * 8;
#pragma unroll
  for (int mt = 0; mt < 4; ++mt) {
#pragma unroll
    for (int nt = 0; nt < 2; ++nt) {
      const int col = n0 + wn * 32 + nt * 16 + (lane & 15);
#pragma unroll
      for (int r = 0; r < 8; ++r) {
        const int row = m0 + wm * 64 + mt * 16 + hi + r;
        const float val = acc[mt][nt][r];
        if constexpr (OUT_F16)
          ((h16*)Cp)[(size_t)row * N + col] = (h16)val;
        else
          ((float*)Cp)[(size_t)row * N + col] = val;
      }
    }
  }
}

// =====================================================================
// Causal flash attention over f16 qkv[T][3][H][DH].
// grid = (T/64, H), block = 128 (4 waves); wave w owns query rows
// q0+16w .. q0+16w+15. Online softmax f32; K/V double-buffered + async DMA.
// =====================================================================
static constexpr int FLD = DH + 8;  // 72

static __device__ __forceinline__ void stage_tile64(h16 (*dst)[FLD],
                                                    const h16* __restrict__ src,
                                                    int tid) {
  // 64x64 halves: 8 chunks/row, 128 thr -> 16 rows/pass, 4 passes (4 async instr)
#pragma unroll
  for (int p = 0; p < 4; ++p) {
    const int r = p * 16 + (tid >> 3);
    const int c = (tid & 7) * 8;
    async_copy16(&dst[r][c], src + (size_t)r * N3 + c);
  }
}

__global__ __launch_bounds__(128) void flash_attn(const h16* __restrict__ qkv,
                                                  h16* __restrict__ y) {
  __shared__ h16 Qs[64][FLD];
  __shared__ h16 Ks[2][64][FLD];
  __shared__ h16 Vs[2][64][FLD];
  __shared__ h16 Ps[4][16][FLD];

  const int tid  = threadIdx.x;
  const int lane = tid & 31;
  const int wave = tid >> 5;
  const int q0   = blockIdx.x * 64;
  const int h    = blockIdx.y;

  stage_tile64(Qs, qkv + (size_t)q0 * N3 + h * DH, tid);                 // 4 ops
  stage_tile64(Ks[0], qkv + (size_t)0 * N3 + D + h * DH, tid);           // 4
  stage_tile64(Vs[0], qkv + (size_t)0 * N3 + 2 * D + h * DH, tid);       // 4
  WAIT_ASYNC_8();   // Q's 4 ops retired (in-order), K/V may still fly
  __syncthreads();

  const v16h qA0 = load_a_frag(&Qs[wave * 16][0], FLD, lane);
  const v16h qA1 = load_a_frag(&Qs[wave * 16][32], FLD, lane);

  const v8f vzero = {};
  v8f o[4];
#pragma unroll
  for (int i = 0; i < 4; ++i) o[i] = vzero;
  float mrow[8], lrow[8];
#pragma unroll
  for (int r = 0; r < 8; ++r) { mrow[r] = -1e30f; lrow[r] = 0.0f; }

  const float kLog2e = 1.44269504f;
  const float kScale = 0.125f;  // 1/sqrt(64)

  int buf = 0;
  for (int kb = 0; kb <= q0; kb += 64) {
    if (kb + 64 <= q0) {  // prefetch next K/V block into the other buffer
      stage_tile64(Ks[buf ^ 1], qkv + (size_t)(kb + 64) * N3 + D + h * DH, tid);
      stage_tile64(Vs[buf ^ 1], qkv + (size_t)(kb + 64) * N3 + 2 * D + h * DH, tid);
      WAIT_ASYNC_8();   // current block's 8 ops retired
    } else {
      WAIT_ASYNC_0();
    }
    __syncthreads();    // K/V[buf] visible to all waves

    // ---- S = Q * K^T : 4 key tiles x (K=64 -> 2 chained WMMA) ----
    v8f s[4];
#pragma unroll
    for (int nt = 0; nt < 4; ++nt) {
      v8f z = vzero;
      z = wmma16(qA0, load_b_fragT(&Ks[buf][nt * 16][0], FLD, lane), z);
      z = wmma16(qA1, load_b_fragT(&Ks[buf][nt * 16][32], FLD, lane), z);
      s[nt] = z;
    }

    // ---- online softmax (per lane: 8 rows, replicated across 16-lane half) ----
    const bool diag = (kb == q0);
#pragma unroll
    for (int r = 0; r < 8; ++r) {
      const int qr = q0 + wave * 16 + r + 8 * (lane >> 4);
      float mx = -1e30f;
#pragma unroll
      for (int nt = 0; nt < 4; ++nt) {
        float v = s[nt][r] * kScale;
        if (diag) {
          const int key = kb + nt * 16 + (lane & 15);
          if (key > qr) v = -1e30f;
        }
        s[nt][r] = v;
        mx = fmaxf(mx, v);
      }
#pragma unroll
      for (int off = 1; off < 16; off <<= 1)
        mx = fmaxf(mx, __shfl_xor(mx, off, 32));
      const float mnew  = fmaxf(mrow[r], mx);
      const float alpha = exp2f((mrow[r] - mnew) * kLog2e);
      float rsum = 0.0f;
#pragma unroll
      for (int nt = 0; nt < 4; ++nt) {
        const float p = exp2f((s[nt][r] - mnew) * kLog2e);
        s[nt][r] = p;
        rsum += p;
      }
#pragma unroll
      for (int off = 1; off < 16; off <<= 1)
        rsum += __shfl_xor(rsum, off, 32);
      lrow[r] = lrow[r] * alpha + rsum;
      mrow[r] = mnew;
#pragma unroll
      for (int nt = 0; nt < 4; ++nt) o[nt][r] *= alpha;
    }

    // ---- stash P (16x64 f16, C-layout -> LDS row-major) ----
#pragma unroll
    for (int nt = 0; nt < 4; ++nt)
#pragma unroll
      for (int r = 0; r < 8; ++r)
        Ps[wave][r + 8 * (lane >> 4)][nt * 16 + (lane & 15)] = (h16)s[nt][r];
    __syncthreads();

    // ---- O += P * V : 4 dh tiles x (K=64 keys -> 2 chained WMMA) ----
    const v16h pA0 = load_a_frag(&Ps[wave][0][0], FLD, lane);
    const v16h pA1 = load_a_frag(&Ps[wave][0][32], FLD, lane);
#pragma unroll
    for (int ntd = 0; ntd < 4; ++ntd) {
      o[ntd] = wmma16(pA0, load_b_frag(&Vs[buf][0][ntd * 16], FLD, lane), o[ntd]);
      o[ntd] = wmma16(pA1, load_b_frag(&Vs[buf][32][ntd * 16], FLD, lane), o[ntd]);
    }

    __syncthreads();    // all waves done reading K/V[buf] before overwrite
    buf ^= 1;
  }

  // ---- normalize and write y[T][D] (f16) ----
#pragma unroll
  for (int r = 0; r < 8; ++r) {
    const float inv = 1.0f / lrow[r];
    const int row = q0 + wave * 16 + r + 8 * (lane >> 4);
#pragma unroll
    for (int ntd = 0; ntd < 4; ++ntd) {
      const int col = h * DH + ntd * 16 + (lane & 15);
      y[(size_t)row * D + col] = (h16)(o[ntd][r] * inv);
    }
  }
}

// =====================================================================
extern "C" void kernel_launch(void* const* d_in, const int* in_sizes, int n_in,
                              void* d_out, int out_size, void* d_ws, size_t ws_size,
                              hipStream_t stream) {
  (void)in_sizes; (void)n_in; (void)out_size; (void)ws_size;
  const float* x    = (const float*)d_in[0];  // [T, D]
  const float* Wqkv = (const float*)d_in[1];  // [D, 3D]
  const float* Wout = (const float*)d_in[2];  // [D, D]
  float* out = (float*)d_out;                 // [T, D] f32

  char* ws = (char*)d_ws;
  h16* xh    = (h16*)ws;                                ws += (size_t)T * D * 2;   // 8 MB
  h16* wqkvh = (h16*)ws;                                ws += (size_t)D * N3 * 2;  // 6 MB
  h16* wouth = (h16*)ws;                                ws += (size_t)D * D * 2;   // 2 MB
  h16* qkv   = (h16*)ws;                                ws += (size_t)T * N3 * 2;  // 24 MB
  h16* yh    = (h16*)ws;                                                            // 8 MB

  // 0) one-shot f32 -> f16 conversions (bandwidth-trivial)
  f32_to_f16<<<dim3((T * D / 4) / 256), dim3(256), 0, stream>>>(x, xh, T * D / 4);
  f32_to_f16<<<dim3((D * N3 / 4) / 256), dim3(256), 0, stream>>>(Wqkv, wqkvh, D * N3 / 4);
  f32_to_f16<<<dim3((D * D / 4) / 256), dim3(256), 0, stream>>>(Wout, wouth, D * D / 4);

  // 1) qkv = xh @ wqkvh   (M=4096, N=3072, K=1024), f16 out
  gemm_wmma<true><<<dim3(N3 / 128, T / 128), dim3(256), 0, stream>>>(
      xh, wqkvh, (void*)qkv, T, N3, D);

  // 2) causal flash attention -> yh (f16)
  flash_attn<<<dim3(T / 64, H), dim3(128), 0, stream>>>(qkv, yh);

  // 3) out = yh @ wouth   (M=4096, N=1024, K=1024), f32 out
  gemm_wmma<false><<<dim3(D / 128, T / 128), dim3(256), 0, stream>>>(
      yh, wouth, (void*)out, T, D, D);
}
